// Layer_80358838108676
// MI455X (gfx1250) — matrix operand order repaired and checked by reading, not compiled, not run on hardware
//
#include <hip/hip_runtime.h>
#include <hip/hip_bf16.h>
#include <math.h>

// ---------------------------------------------------------------------------
// MI455X / gfx1250 implementation. wave32, WMMA bf16 (16x16x32) for all GEMMs.
// ---------------------------------------------------------------------------

typedef __bf16 bf16;
typedef __attribute__((ext_vector_type(16))) bf16  v16bf;
typedef __attribute__((ext_vector_type(8)))  bf16  v8bf;
typedef __attribute__((ext_vector_type(8)))  float v8f;

#define BIGW 1e30f

static constexpr int Bb  = 8;
static constexpr int HH  = 64;
static constexpr int WW  = 64;
static constexpr int LL  = HH * WW;      // 4096
static constexpr int DM  = 128;
static constexpr int DI  = 256;
static constexpr int HID = 512;
static constexpr int NROW = Bb * LL;     // 32768

__device__ __forceinline__ float wred_sum(float v) {
#pragma unroll
  for (int m = 16; m; m >>= 1) v += __shfl_xor(v, m, 32);
  return v;
}
__device__ __forceinline__ float siluf(float x) { return x / (1.f + expf(-x)); }
__device__ __forceinline__ float geluf(float x) {
  float x3 = x * x * x;
  return 0.5f * x * (1.f + tanhf(0.7978845608028654f * (x + 0.044715f * x3)));
}

// ---------------------------------------------------------------- f32 -> bf16
__global__ void cvt_bf16_kernel(const float* __restrict__ in,
                                bf16* __restrict__ out, int n) {
  int i = blockIdx.x * 256 + threadIdx.x;
  if (i < n) out[i] = (bf16)in[i];
}

// ------------------------------------------------------- LayerNorm over 128
// one wave per row; 4 elems / lane
__global__ __launch_bounds__(256) void ln128_kernel(
    const float* __restrict__ x, const float* __restrict__ g,
    const float* __restrict__ b, bf16* __restrict__ out, float eps) {
  int row  = blockIdx.x * 8 + (threadIdx.x >> 5);
  int lane = threadIdx.x & 31;
  const float* xr = x + (size_t)row * DM;
  float v[4], s = 0.f;
#pragma unroll
  for (int i = 0; i < 4; ++i) { v[i] = xr[lane + 32 * i]; s += v[i]; }
  float mu = wred_sum(s) * (1.f / DM);
  float q = 0.f;
#pragma unroll
  for (int i = 0; i < 4; ++i) { float d = v[i] - mu; q += d * d; }
  float rs = rsqrtf(wred_sum(q) * (1.f / DM) + eps);
#pragma unroll
  for (int i = 0; i < 4; ++i) {
    int c = lane + 32 * i;
    out[(size_t)row * DM + c] = (bf16)((v[i] - mu) * rs * g[c] + b[c]);
  }
}

// --------------------------------------------------------------- WMMA GEMM
// C[M,N] = A[M,K] * W[N,K]^T  (A,W bf16 row-major, K-contiguous; f32 accum)
// block = 8 waves stacked along M (tile 128 x 16). EPI selects fused epilogue.
//  EPI 0: split: cols<256 -> xc channel-first (o0); cols>=256 -> silu -> z (o1)
//  EPI 1: o0 = C + p0 (residual, N=128)
//  EPI 2: ob = gelu(C + p0[j]) as bf16 (N=512)
//  EPI 3: o0 = C + p0[j] + p1 (bias + residual, N=128)
template <int EPI>
__global__ __launch_bounds__(256) void gemm_wmma_kernel(
    const bf16* __restrict__ A, const bf16* __restrict__ Wt, int K,
    const float* __restrict__ p0, const float* __restrict__ p1,
    float* __restrict__ o0, float* __restrict__ o1, bf16* __restrict__ ob) {
  const int lane = threadIdx.x & 31;
  const int wave = threadIdx.x >> 5;
  const int m0   = blockIdx.y * 128 + wave * 16;
  const int n0   = blockIdx.x * 16;
  const int half = lane >> 4;     // 0: lanes 0-15, 1: lanes 16-31
  const int l15  = lane & 15;

  const bf16* __restrict__ arow = A  + (size_t)(m0 + l15) * K + half * 16;
  const bf16* __restrict__ brow = Wt + (size_t)(n0 + l15) * K + half * 16;

  v8f acc = {};
  for (int k0 = 0; k0 < K; k0 += 32) {
    v8bf  alo = *(const v8bf*)(arow + k0);
    v8bf  ahi = *(const v8bf*)(arow + k0 + 8);
    v16bf bf_ = *(const v16bf*)(brow + k0);
    if (k0 + 32 < K) {
      __builtin_prefetch(arow + k0 + 32, 0, 1);   // global_prefetch_b8
      __builtin_prefetch(brow + k0 + 32, 0, 1);
    }
    v16bf af;
#pragma unroll
    for (int i = 0; i < 8; ++i) { af[i] = alo[i]; af[i + 8] = ahi[i]; }
    acc = __builtin_amdgcn_wmma_f32_16x16x32_bf16(
        /*neg_a=*/false, af, /*neg_b=*/false, bf_,
        /*c_mod=*/(short)0, acc, /*reuse_a=*/false, /*reuse_b=*/false);
  }

  const int j = n0 + l15;
#pragma unroll
  for (int r = 0; r < 8; ++r) {
    int gr = m0 + r + half * 8;          // global row (b*L + l)
    float val = acc[r];
    if constexpr (EPI == 0) {
      int bi = gr >> 12, l = gr & (LL - 1);
      if (j < DI) o0[((size_t)bi * DI + j) * LL + l] = val;        // xc (NCHW)
      else        o1[(size_t)gr * DI + (j - DI)]    = siluf(val);  // z
    } else if constexpr (EPI == 1) {
      o0[(size_t)gr * DM + j] = val + p0[(size_t)gr * DM + j];
    } else if constexpr (EPI == 2) {
      ob[(size_t)gr * HID + j] = (bf16)geluf(val + p0[j]);
    } else {
      o0[(size_t)gr * DM + j] = val + p0[j] + p1[(size_t)gr * DM + j];
    }
  }
}

// ----------------------------------------------- depthwise 3x3 conv + SiLU
__global__ __launch_bounds__(256) void dwconv_kernel(
    const float* __restrict__ xc, const float* __restrict__ w,
    float* __restrict__ xs) {
  int idx = blockIdx.x * 256 + threadIdx.x;         // b*DI*L
  int l = idx & (LL - 1);
  int c = (idx >> 12) & (DI - 1);
  int b = idx >> 20;
  int r = l >> 6, col = l & 63;
  const float* base = xc + ((size_t)b * DI + c) * LL;
  const float* wc   = w + c * 9;
  float s = 0.f;
#pragma unroll
  for (int ky = 0; ky < 3; ++ky) {
    int yy = r + ky - 1;
    if (yy < 0 || yy >= HH) continue;
#pragma unroll
    for (int kx = 0; kx < 3; ++kx) {
      int xx = col + kx - 1;
      if (xx < 0 || xx >= WW) continue;
      s += wc[ky * 3 + kx] * base[yy * WW + xx];
    }
  }
  xs[((size_t)b * DI + c) * LL + l] = siluf(s);
}

// --------------- x_dbl (DI->10) + dt (8->DI) + softplus + deltaA + feat_in
// thread per (b,l); weights staged in LDS; outputs channel-LAST (B,L,DI)
__global__ __launch_bounds__(256) void ssm_in_kernel(
    const float* __restrict__ xs, const float* __restrict__ xw,
    const float* __restrict__ dtw, const float* __restrict__ dtb,
    const float* __restrict__ alog, float* __restrict__ fi,
    float* __restrict__ dA, float* __restrict__ Cs) {
  __shared__ float sxw[10 * DI];
  __shared__ float sdtw[DI * 8];
  __shared__ float sbias[DI];
  __shared__ float sA[DI];
  int tid = threadIdx.x;
  for (int i = tid; i < 10 * DI; i += 256) sxw[i] = xw[i];
  for (int i = tid; i < DI * 8; i += 256)  sdtw[i] = dtw[i];
  sbias[tid] = dtb[tid];
  sA[tid]    = -expf(alog[tid]);
  __syncthreads();

  int row = blockIdx.x * 256 + tid;        // b*L + l
  int b = row >> 12, l = row & (LL - 1);
  const float* xcol = xs + (size_t)b * DI * LL + l;

  float acc[10];
#pragma unroll
  for (int c = 0; c < 10; ++c) acc[c] = 0.f;
  for (int d = 0; d < DI; ++d) {
    float xv = xcol[(size_t)d * LL];
#pragma unroll
    for (int c = 0; c < 10; ++c) acc[c] += sxw[c * DI + d] * xv;
  }
  float Bsv = acc[8];
  Cs[row]   = acc[9];

  size_t ob = (size_t)row * DI;
  for (int d = 0; d < DI; ++d) {
    float t = sbias[d];
#pragma unroll
    for (int r = 0; r < 8; ++r) t += sdtw[d * 8 + r] * acc[r];
    float sp = (t > 20.f) ? t : log1pf(expf(t));
    float xv = xcol[(size_t)d * LL];
    fi[ob + d] = sp * Bsv * xv;
    dA[ob + d] = expf(sp * sA[d]);
  }
}

// ------------------------------------------------ grid weights wr, wd
__global__ __launch_bounds__(256) void gridw_kernel(
    const float* __restrict__ xs, float* __restrict__ wr,
    float* __restrict__ wd) {
  int idx = blockIdx.x * 256 + threadIdx.x;    // b*L
  int b = idx >> 12, l = idx & (LL - 1);
  int r = l >> 6, c = l & 63;
  const float* base = xs + (size_t)b * DI * LL;
  bool hasR = (c < WW - 1), hasD = (r < HH - 1);
  float dotr = 0, dotd = 0, na = 0, nbr = 0, nbd = 0;
  for (int d = 0; d < DI; ++d) {
    float a = base[(size_t)d * LL + l];
    na += a * a;
    if (hasR) { float x = base[(size_t)d * LL + l + 1];  dotr += a * x; nbr += x * x; }
    if (hasD) { float x = base[(size_t)d * LL + l + WW]; dotd += a * x; nbd += x * x; }
  }
  wr[idx] = hasR ? expf(-dotr / fmaxf(sqrtf(na) * sqrtf(nbr), 1e-8f)) : BIGW;
  wd[idx] = hasD ? expf(-dotd / fmaxf(sqrtf(na) * sqrtf(nbd), 1e-8f)) : BIGW;
}

// ------------------------------------------------ Prim's MST (LDS-resident)
__global__ __launch_bounds__(256) void mst_kernel(
    const float* __restrict__ wrg, const float* __restrict__ wdg,
    int* __restrict__ orderg, int* __restrict__ parg) {
  __shared__ float sdist[LL];
  __shared__ int   spar[LL];
  __shared__ unsigned char sin_[LL];
  __shared__ float rv[256];
  __shared__ int   ri[256];
  int b = blockIdx.x, tid = threadIdx.x;
  const float* wr = wrg + (size_t)b * LL;
  const float* wd = wdg + (size_t)b * LL;
  for (int i = tid; i < LL; i += 256) { sdist[i] = BIGW; spar[i] = 0; sin_[i] = 0; }
  __syncthreads();
  if (tid == 0) sdist[0] = -1.f;
  __syncthreads();

  for (int t = 0; t < LL; ++t) {
    float bv = 3.4e38f; int bi = LL;
    for (int i = tid; i < LL; i += 256) {
      float v = sin_[i] ? BIGW : sdist[i];
      if (v < bv || (v == bv && i < bi)) { bv = v; bi = i; }
    }
    rv[tid] = bv; ri[tid] = bi;
    __syncthreads();
    for (int s = 128; s; s >>= 1) {
      if (tid < s) {
        if (rv[tid + s] < rv[tid] ||
            (rv[tid + s] == rv[tid] && ri[tid + s] < ri[tid])) {
          rv[tid] = rv[tid + s]; ri[tid] = ri[tid + s];
        }
      }
      __syncthreads();
    }
    int v = ri[0];
    if (tid == 0) {
      sin_[v] = 1;
      orderg[(size_t)b * LL + t] = v;
      int r = v >> 6, c = v & 63;
      if (c < WW - 1) { int u = v + 1;  float w = wr[v];      if (!sin_[u] && w < sdist[u]) { sdist[u] = w; spar[u] = v; } }
      if (c > 0)      { int u = v - 1;  float w = wr[v - 1];  if (!sin_[u] && w < sdist[u]) { sdist[u] = w; spar[u] = v; } }
      if (r < HH - 1) { int u = v + WW; float w = wd[v];      if (!sin_[u] && w < sdist[u]) { sdist[u] = w; spar[u] = v; } }
      if (r > 0)      { int u = v - WW; float w = wd[v - WW]; if (!sin_[u] && w < sdist[u]) { sdist[u] = w; spar[u] = v; } }
    }
    __syncthreads();
  }
  for (int i = tid; i < LL; i += 256) parg[(size_t)b * LL + i] = spar[i];
}

// ------------------------------- tree up-sweep + down-sweep (channel-last)
// thread = channel; per-channel chains are independent -> no barriers in loop
__global__ __launch_bounds__(256) void treescan_kernel(
    float* __restrict__ up, const float* __restrict__ ew,
    const int* __restrict__ orderg, const int* __restrict__ parg,
    float* __restrict__ out) {
  __shared__ int so[LL];
  __shared__ int sp[LL];
  int b = blockIdx.x, tid = threadIdx.x;
  for (int i = tid; i < LL; i += 256) {
    so[i] = orderg[(size_t)b * LL + i];
    sp[i] = parg[(size_t)b * LL + i];
  }
  __syncthreads();
  size_t base = (size_t)b * LL * DI;
  int ch = tid;
  for (int t = LL - 1; t >= 1; --t) {
    int v = so[t], p = sp[v];
    float w  = ew[base + (size_t)v * DI + ch];
    float uv = up[base + (size_t)v * DI + ch];
    up[base + (size_t)p * DI + ch] += w * uv;
  }
  int root = so[0];
  out[base + (size_t)root * DI + ch] = up[base + (size_t)root * DI + ch];
  for (int t = 1; t < LL; ++t) {
    int v = so[t], p = sp[v];
    float w  = ew[base + (size_t)v * DI + ch];
    float uv = up[base + (size_t)v * DI + ch];
    float op = out[base + (size_t)p * DI + ch];
    out[base + (size_t)v * DI + ch] = uv + w * (op - w * uv);
  }
}

// ----------------- hn-LN, *Cs, +ds*xs, on-LN, *z  -> g (bf16) for w_out GEMM
__global__ __launch_bounds__(256) void post_kernel(
    const float* __restrict__ fo, const float* __restrict__ Cs,
    const float* __restrict__ xs, const float* __restrict__ z,
    const float* __restrict__ dsv, const float* __restrict__ hng,
    const float* __restrict__ hnb, const float* __restrict__ ong,
    const float* __restrict__ onb, bf16* __restrict__ g) {
  int row  = blockIdx.x * 8 + (threadIdx.x >> 5);
  int lane = threadIdx.x & 31;
  int b = row >> 12, l = row & (LL - 1);
  const float* fr = fo + (size_t)row * DI;
  float f[8], s = 0.f;
#pragma unroll
  for (int i = 0; i < 8; ++i) { f[i] = fr[lane + 32 * i]; s += f[i]; }
  float mu = wred_sum(s) * (1.f / DI);
  float q = 0.f;
#pragma unroll
  for (int i = 0; i < 8; ++i) { float d = f[i] - mu; q += d * d; }
  float rs = rsqrtf(wred_sum(q) * (1.f / DI) + 1e-5f);
  float cv = Cs[row];
  float y[8]; s = 0.f;
#pragma unroll
  for (int i = 0; i < 8; ++i) {
    int c = lane + 32 * i;
    float ho = (f[i] - mu) * rs * hng[c] + hnb[c];
    y[i] = ho * cv + dsv[c] * xs[((size_t)b * DI + c) * LL + l];
    s += y[i];
  }
  float mu2 = wred_sum(s) * (1.f / DI);
  q = 0.f;
#pragma unroll
  for (int i = 0; i < 8; ++i) { float d = y[i] - mu2; q += d * d; }
  float rs2 = rsqrtf(wred_sum(q) * (1.f / DI) + 1e-5f);
#pragma unroll
  for (int i = 0; i < 8; ++i) {
    int c = lane + 32 * i;
    float yy = (y[i] - mu2) * rs2 * ong[c] + onb[c];
    g[(size_t)row * DI + c] = (bf16)(yy * z[(size_t)row * DI + c]);
  }
}

// ===========================================================================
extern "C" void kernel_launch(void* const* d_in, const int* in_sizes, int n_in,
                              void* d_out, int out_size, void* d_ws,
                              size_t ws_size, hipStream_t stream) {
  const float* x      = (const float*)d_in[0];
  const float* ln1_g  = (const float*)d_in[1];
  const float* ln1_b  = (const float*)d_in[2];
  const float* ln2_g  = (const float*)d_in[3];
  const float* ln2_b  = (const float*)d_in[4];
  const float* w_in   = (const float*)d_in[5];
  const float* conv_w = (const float*)d_in[6];
  const float* x_w    = (const float*)d_in[7];
  const float* dt_w   = (const float*)d_in[8];
  const float* dt_b   = (const float*)d_in[9];
  const float* a_logs = (const float*)d_in[10];
  const float* ds_    = (const float*)d_in[11];
  const float* hn_g   = (const float*)d_in[12];
  const float* hn_b   = (const float*)d_in[13];
  const float* on_g   = (const float*)d_in[14];
  const float* on_b   = (const float*)d_in[15];
  const float* w_out  = (const float*)d_in[16];
  const float* mlp_w1 = (const float*)d_in[17];
  const float* mlp_b1 = (const float*)d_in[18];
  const float* mlp_w2 = (const float*)d_in[19];
  const float* mlp_b2 = (const float*)d_in[20];
  float* out = (float*)d_out;

  // ---- workspace carving (256B aligned) ----
  char* p = (char*)d_ws;
  size_t off = 0;
  auto carve = [&](size_t bytes) {
    void* r = p + off;
    off += (bytes + 255) & ~(size_t)255;
    return r;
  };
  bf16*  hbf    = (bf16*)carve((size_t)NROW * DM * 2);   // LN1 out; reused as h2
  bf16*  winbf  = (bf16*)carve((size_t)2 * DI * DM * 2);
  bf16*  woutbf = (bf16*)carve((size_t)DM * DI * 2);
  bf16*  w1bf   = (bf16*)carve((size_t)HID * DM * 2);
  bf16*  w2bf   = (bf16*)carve((size_t)DM * HID * 2);
  float* xcfo   = (float*)carve((size_t)Bb * DI * LL * 4); // xc (NCHW) -> feat_out
  float* zbuf   = (float*)carve((size_t)NROW * DI * 4);
  float* xsbuf  = (float*)carve((size_t)Bb * DI * LL * 4);
  float* upbuf  = (float*)carve((size_t)NROW * DI * 4);    // feat_in / up
  float* dAbuf  = (float*)carve((size_t)NROW * DI * 4);    // deltaA -> abf (bf16)
  float* Csbuf  = (float*)carve((size_t)NROW * 4);
  float* wrbuf  = (float*)carve((size_t)Bb * LL * 4);
  float* wdbuf  = (float*)carve((size_t)Bb * LL * 4);
  int*   ordbuf = (int*)carve((size_t)Bb * LL * 4);
  int*   parbuf = (int*)carve((size_t)Bb * LL * 4);
  bf16*  gbf    = (bf16*)carve((size_t)NROW * DI * 2);
  float* x1buf  = (float*)carve((size_t)NROW * DM * 4);
  bf16*  abf    = (bf16*)dAbuf;  // alias: deltaA dead after tree scan
  (void)in_sizes; (void)n_in; (void)out_size; (void)ws_size;

  // ---- weight conversion to bf16 ----
  cvt_bf16_kernel<<<(2 * DI * DM + 255) / 256, 256, 0, stream>>>(w_in, winbf, 2 * DI * DM);
  cvt_bf16_kernel<<<(DM * DI + 255) / 256, 256, 0, stream>>>(w_out, woutbf, DM * DI);
  cvt_bf16_kernel<<<(HID * DM + 255) / 256, 256, 0, stream>>>(mlp_w1, w1bf, HID * DM);
  cvt_bf16_kernel<<<(DM * HID + 255) / 256, 256, 0, stream>>>(mlp_w2, w2bf, DM * HID);

  // ---- LN1 ----
  ln128_kernel<<<NROW / 8, 256, 0, stream>>>(x, ln1_g, ln1_b, hbf, 1e-6f);

  // ---- GEMM1: h @ w_in^T, split into xc (NCHW) and silu(z) ----
  gemm_wmma_kernel<0><<<dim3((2 * DI) / 16, NROW / 128), 256, 0, stream>>>(
      hbf, winbf, DM, nullptr, nullptr, xcfo, zbuf, nullptr);

  // ---- depthwise conv + silu -> xs ----
  dwconv_kernel<<<(Bb * DI * LL) / 256, 256, 0, stream>>>(xcfo, conv_w, xsbuf);

  // ---- x_dbl / dt / softplus / deltaA / feat_in (channel-last) ----
  ssm_in_kernel<<<NROW / 256, 256, 0, stream>>>(
      xsbuf, x_w, dt_w, dt_b, a_logs, upbuf, dAbuf, Csbuf);

  // ---- grid weights ----
  gridw_kernel<<<(Bb * LL) / 256, 256, 0, stream>>>(xsbuf, wrbuf, wdbuf);

  // ---- MST (one WGP block per batch, LDS-resident state) ----
  mst_kernel<<<Bb, 256, 0, stream>>>(wrbuf, wdbuf, ordbuf, parbuf);

  // ---- tree scan (up + down) ----
  treescan_kernel<<<Bb, 256, 0, stream>>>(upbuf, dAbuf, ordbuf, parbuf, xcfo);

  // ---- hn-LN, gate, on-LN, *z -> g(bf16) ----
  post_kernel<<<NROW / 8, 256, 0, stream>>>(
      xcfo, Csbuf, xsbuf, zbuf, ds_, hn_g, hn_b, on_g, on_b, gbf);

  // ---- GEMM3: g @ w_out^T + x -> x1 ----
  gemm_wmma_kernel<1><<<dim3(DM / 16, NROW / 128), 256, 0, stream>>>(
      gbf, woutbf, DI, x, nullptr, x1buf, nullptr, nullptr);

  // ---- LN2 ----
  ln128_kernel<<<NROW / 8, 256, 0, stream>>>(x1buf, ln2_g, ln2_b, hbf, 1e-6f);

  // ---- MLP1: gelu(h2 @ w1^T + b1) -> abf ----
  gemm_wmma_kernel<2><<<dim3(HID / 16, NROW / 128), 256, 0, stream>>>(
      hbf, w1bf, DM, mlp_b1, nullptr, nullptr, nullptr, abf);

  // ---- MLP2: a @ w2^T + b2 + x1 -> out ----
  gemm_wmma_kernel<3><<<dim3(DM / 16, NROW / 128), 256, 0, stream>>>(
      abf, w2bf, HID, mlp_b2, x1buf, out, nullptr, nullptr);
}
